// MiMoAudioDecoderLayer_77738908057994
// MI455X (gfx1250) — matrix operand order repaired and checked by reading, not compiled, not run on hardware
//
#include <hip/hip_runtime.h>

// ---------------------------------------------------------------------------
// MiMoAudio decoder layer for gfx1250 (MI455X), bf16 WMMA path.
// B=2 S=2048 H=2048 NH=16 NKV=4 D=128 INTER=8192, fp32 in/out.
// ---------------------------------------------------------------------------

#define BB     2
#define SS     2048
#define HH     2048
#define NH_    16
#define NKV_   4
#define DD     128
#define INTER_ 8192
#define MTOT   (BB * SS)          // 4096 rows of tokens
#define NQKV   (NH_ * DD + 2 * NKV_ * DD)  // 3072

typedef __attribute__((ext_vector_type(16))) __bf16 v16bf;
typedef __attribute__((ext_vector_type(8)))  float  v8f;
typedef __attribute__((ext_vector_type(4)))  int    v4i;

union F16x16 { v16bf v; uint4 q[2]; };   // 32 bytes: one WMMA 16-bit A/B fragment per lane

__device__ inline v8f v8f_zero() {
  v8f z;
#pragma unroll
  for (int i = 0; i < 8; ++i) z[i] = 0.f;
  return z;
}

// ---------------------------------------------------------------------------
// CDNA5 async global->LDS copy path (ASYNCcnt-tracked copies; enables a
// single-barrier double-buffered GEMM K-loop).
// ---------------------------------------------------------------------------
#if defined(__has_builtin)
#  if __has_builtin(__builtin_amdgcn_global_load_async_to_lds_b128)
#    define USE_ASYNC_COPY 1
#  endif
#endif
#ifndef USE_ASYNC_COPY
#  define USE_ASYNC_COPY 0
#endif

#if USE_ASYNC_COPY
__device__ __forceinline__ void async_b128(const void* g, void* l) {
  // per-lane: LDS[l] = GLOBAL[g], 16 bytes, tracked by ASYNCcnt
  __builtin_amdgcn_global_load_async_to_lds_b128(
      (__attribute__((address_space(1))) v4i*)g,
      (__attribute__((address_space(3))) v4i*)l, 0, 0);
}
__device__ __forceinline__ void wait_async0() {
#  if __has_builtin(__builtin_amdgcn_s_wait_asynccnt)
  __builtin_amdgcn_s_wait_asynccnt(0);
#  else
  asm volatile("s_wait_asynccnt 0x0" ::: "memory");
#  endif
}
#endif

// ---------------------------------------------------------------------------
// Weight transpose + fp32 -> bf16 convert.  W[K][N] fp32  ->  Wt[N][K] bf16.
// K, N multiples of 32.
// ---------------------------------------------------------------------------
__global__ __launch_bounds__(256) void k_transpose_bf16(
    const float* __restrict__ W, __bf16* __restrict__ Wt, int K, int N) {
  __shared__ float tile[32][33];
  const int n0 = blockIdx.x * 32;
  const int k0 = blockIdx.y * 32;
  const int tx = threadIdx.x;   // 0..31
  const int ty = threadIdx.y;   // 0..7
#pragma unroll
  for (int i = 0; i < 32; i += 8)
    tile[ty + i][tx] = W[(size_t)(k0 + ty + i) * N + (n0 + tx)];
  __syncthreads();
#pragma unroll
  for (int i = 0; i < 32; i += 8)
    Wt[(size_t)(n0 + ty + i) * K + (k0 + tx)] = (__bf16)tile[tx][ty + i];
}

// ---------------------------------------------------------------------------
// RMSNorm: fp32 row -> bf16 row.  One 256-thread block per row, H columns.
// ---------------------------------------------------------------------------
__global__ __launch_bounds__(256) void k_rmsnorm(
    const float* __restrict__ x, const float* __restrict__ w,
    __bf16* __restrict__ out, int H) {
  const int row = blockIdx.x;
  const int tid = threadIdx.x;
  const float* xr = x + (size_t)row * H;
  float ss = 0.f;
  for (int i = tid; i < H; i += 256) { float v = xr[i]; ss += v * v; }
#pragma unroll
  for (int off = 16; off >= 1; off >>= 1) ss += __shfl_xor(ss, off, 32);
  __shared__ float red[8];
  if ((tid & 31) == 0) red[tid >> 5] = ss;
  __syncthreads();
  float tot = 0.f;
#pragma unroll
  for (int i = 0; i < 8; ++i) tot += red[i];
  const float rstd = rsqrtf(tot / (float)H + 1e-6f);
  for (int i = tid; i < H; i += 256)
    out[(size_t)row * H + i] = (__bf16)(xr[i] * rstd * w[i]);
}

// ---------------------------------------------------------------------------
// bf16 WMMA GEMM:  C[M][N] (+epilogue) = A[M][K] @ Bt[N][K]^T
// Block tile 128x128, K-step 32, 8 waves (2x4), each wave 64x32 via 4x2
// v_wmma_f32_16x16x32_bf16.  M,N multiples of 128; K multiple of 32.
// Async path: double-buffered LDS, one barrier per K-step, fills via
// global_load_async_to_lds_b128.  Fallback: sync copy, two barriers.
// epi: 0 = store f32
//      1 = store f32  (extra_f32 + acc)          [residual add]
//      2 = store bf16 silu(acc)                  [gate]
//      3 = store bf16 (acc * (f32)extra_bf16)    [up * silu(gate)]
// ---------------------------------------------------------------------------
__global__ __launch_bounds__(256) void k_gemm(
    const __bf16* __restrict__ A, const __bf16* __restrict__ Bt,
    void* __restrict__ Cout, const void* __restrict__ extra,
    int M, int N, int K, int epi) {
  const int tid  = threadIdx.x;
  const int lane = tid & 31;
  const int wave = tid >> 5;
  const int wm = wave >> 2;          // 0..1  (rows of waves)
  const int wn = wave & 3;           // 0..3  (cols of waves)
  const int nn = lane & 15;          // lane within half
  const int hi = lane >> 4;          // half-wave select
  const int row0 = blockIdx.y * 128;
  const int col0 = blockIdx.x * 128;

  v8f acc[4][2];
#pragma unroll
  for (int i = 0; i < 4; ++i)
#pragma unroll
    for (int j = 0; j < 2; ++j) acc[i][j] = v8f_zero();

  const int lr = tid >> 1;           // 0..127 : LDS row loaded by this thread
  const int lh = tid & 1;            // 0..1   : 16-element half of the K-step

#if USE_ASYNC_COPY
  __shared__ alignas(16) __bf16 As[2][128][40];   // padded 80B rows, 16B aligned
  __shared__ alignas(16) __bf16 Bs[2][128][40];

  // prologue: start tile kk=0 into buffer 0
  {
    const __bf16* ga = A  + (size_t)(row0 + lr) * K + lh * 16;
    const __bf16* gb = Bt + (size_t)(col0 + lr) * K + lh * 16;
    async_b128(ga,     &As[0][lr][lh * 16]);
    async_b128(ga + 8, &As[0][lr][lh * 16 + 8]);
    async_b128(gb,     &Bs[0][lr][lh * 16]);
    async_b128(gb + 8, &Bs[0][lr][lh * 16 + 8]);
  }
  int pp = 0;
  for (int kk = 0; kk < K; kk += 32, pp ^= 1) {
    wait_async0();          // this wave's fills done
    __syncthreads();        // everyone's fills done; prior reads of pp^1 done
    if (kk + 32 < K) {      // start next tile into the other buffer
      const __bf16* ga = A  + (size_t)(row0 + lr) * K + (kk + 32) + lh * 16;
      const __bf16* gb = Bt + (size_t)(col0 + lr) * K + (kk + 32) + lh * 16;
      async_b128(ga,     &As[pp ^ 1][lr][lh * 16]);
      async_b128(ga + 8, &As[pp ^ 1][lr][lh * 16 + 8]);
      async_b128(gb,     &Bs[pp ^ 1][lr][lh * 16]);
      async_b128(gb + 8, &Bs[pp ^ 1][lr][lh * 16 + 8]);
    }
    F16x16 af[4], bfr[2];
#pragma unroll
    for (int i = 0; i < 4; ++i) {
      const int m = wm * 64 + i * 16 + nn;
      af[i].q[0] = *(const uint4*)&As[pp][m][hi * 8];
      af[i].q[1] = *(const uint4*)&As[pp][m][hi * 8 + 16];
    }
#pragma unroll
    for (int j = 0; j < 2; ++j) {
      const int n = wn * 32 + j * 16 + nn;
      bfr[j].q[0] = *(const uint4*)&Bs[pp][n][hi * 8];
      bfr[j].q[1] = *(const uint4*)&Bs[pp][n][hi * 8 + 16];
    }
#pragma unroll
    for (int i = 0; i < 4; ++i)
#pragma unroll
      for (int j = 0; j < 2; ++j)
        acc[i][j] = __builtin_amdgcn_wmma_f32_16x16x32_bf16(
            false, af[i].v, false, bfr[j].v, (short)0, acc[i][j], false, false);
  }
#else
  __shared__ alignas(16) __bf16 As[128][40];
  __shared__ alignas(16) __bf16 Bs[128][40];

  for (int kk = 0; kk < K; kk += 32) {
    const uint4* ga = (const uint4*)(A  + (size_t)(row0 + lr) * K + kk + lh * 16);
    const uint4* gb = (const uint4*)(Bt + (size_t)(col0 + lr) * K + kk + lh * 16);
    uint4 a0 = ga[0], a1 = ga[1];
    uint4 b0 = gb[0], b1 = gb[1];
    *(uint4*)&As[lr][lh * 16]     = a0;
    *(uint4*)&As[lr][lh * 16 + 8] = a1;
    *(uint4*)&Bs[lr][lh * 16]     = b0;
    *(uint4*)&Bs[lr][lh * 16 + 8] = b1;
    __syncthreads();

    F16x16 af[4], bfr[2];
#pragma unroll
    for (int i = 0; i < 4; ++i) {
      const int m = wm * 64 + i * 16 + nn;
      af[i].q[0] = *(const uint4*)&As[m][hi * 8];
      af[i].q[1] = *(const uint4*)&As[m][hi * 8 + 16];
    }
#pragma unroll
    for (int j = 0; j < 2; ++j) {
      const int n = wn * 32 + j * 16 + nn;
      bfr[j].q[0] = *(const uint4*)&Bs[n][hi * 8];
      bfr[j].q[1] = *(const uint4*)&Bs[n][hi * 8 + 16];
    }
#pragma unroll
    for (int i = 0; i < 4; ++i)
#pragma unroll
      for (int j = 0; j < 2; ++j)
        acc[i][j] = __builtin_amdgcn_wmma_f32_16x16x32_bf16(
            false, af[i].v, false, bfr[j].v, (short)0, acc[i][j], false, false);
    __syncthreads();
  }
#endif

  // Epilogue.  C/D layout: lanes<16 -> M=r, lanes>=16 -> M=8+r; N = lane&15.
#pragma unroll
  for (int i = 0; i < 4; ++i)
#pragma unroll
    for (int j = 0; j < 2; ++j)
#pragma unroll
      for (int r = 0; r < 8; ++r) {
        const int m = row0 + wm * 64 + i * 16 + hi * 8 + r;
        const int n = col0 + wn * 32 + j * 16 + nn;
        const size_t idx = (size_t)m * N + n;
        const float v = acc[i][j][r];
        if (epi == 0) {
          ((float*)Cout)[idx] = v;
        } else if (epi == 1) {
          ((float*)Cout)[idx] = ((const float*)extra)[idx] + v;
        } else if (epi == 2) {
          ((__bf16*)Cout)[idx] = (__bf16)(v / (1.f + __expf(-v)));
        } else {
          const float g = (float)((const __bf16*)extra)[idx];
          ((__bf16*)Cout)[idx] = (__bf16)(v * g);
        }
      }
}

// ---------------------------------------------------------------------------
// RoPE + bias: read fp32 qkv rows, write bf16 [b,s,nheads,D].
// colOff selects q (0) or k (2048) within the fused qkv buffer (ldc=3072).
// ---------------------------------------------------------------------------
__global__ __launch_bounds__(256) void k_rope(
    const float* __restrict__ qkv, const float* __restrict__ bias,
    const int* __restrict__ pos, __bf16* __restrict__ outb,
    int nheads, int colOff) {
  const int total = BB * SS * nheads * (DD / 2);
  int idx = blockIdx.x * 256 + threadIdx.x;
  if (idx >= total) return;
  const int d2 = idx & 63;  idx >>= 6;
  const int h  = idx % nheads;  idx /= nheads;
  const int s  = idx % SS;
  const int b  = idx / SS;
  const float p   = (float)pos[s];
  const float inv = __powf(1000000.f, -(float)d2 / 64.f);
  float sn, c;
  __sincosf(p * inv, &sn, &c);
  const size_t rb = (size_t)(b * SS + s) * NQKV + colOff + h * DD;
  const float x1 = qkv[rb + d2]      + bias[h * DD + d2];
  const float x2 = qkv[rb + 64 + d2] + bias[h * DD + 64 + d2];
  const size_t ob = ((size_t)(b * SS + s) * nheads + h) * DD;
  outb[ob + d2]      = (__bf16)(x1 * c - x2 * sn);
  outb[ob + 64 + d2] = (__bf16)(x2 * c + x1 * sn);
}

// V + bias, transposed to [b, hkv, D, S] bf16 so the P@V B-fragment
// (two runs of 8 contiguous keys at fixed d-column) is two 16B loads.
__global__ __launch_bounds__(256) void k_vtrans(
    const float* __restrict__ qkv, const float* __restrict__ bv,
    __bf16* __restrict__ vt) {
  const int total = BB * SS * NKV_ * DD;
  int idx = blockIdx.x * 256 + threadIdx.x;
  if (idx >= total) return;
  const int d = idx & (DD - 1);  idx >>= 7;
  const int h = idx % NKV_;  idx /= NKV_;
  const int s = idx % SS;
  const int b = idx / SS;
  const float v = qkv[(size_t)(b * SS + s) * NQKV + (NH_ * DD + NKV_ * DD) + h * DD + d]
                + bv[h * DD + d];
  vt[((size_t)((b * NKV_ + h) * DD + d)) * SS + s] = (__bf16)v;
}

// ---------------------------------------------------------------------------
// Causal GQA flash attention.  One wave per 16-row Q tile, 4 waves/block.
// Q bf16 [b,s,NH,D]; K bf16 [b,s,NKV,D]; Vt bf16 [b,hkv,D,S]; out bf16 [b,s,NH*D].
// Per 32-key step: 8x QK^T wmma + 8x P@V wmma; online softmax via half-wave
// shfl_xor reductions; P transposed C->A layout through per-wave LDS.
// ---------------------------------------------------------------------------
#define WPB_ATT 4
__global__ __launch_bounds__(32 * WPB_ATT) void k_attention(
    const __bf16* __restrict__ qb, const __bf16* __restrict__ kb,
    const __bf16* __restrict__ vt, __bf16* __restrict__ ob) {
  const int lane = threadIdx.x & 31;
  const int wave = threadIdx.x >> 5;
  const int qt   = blockIdx.x * WPB_ATT + wave;
  const int bh   = blockIdx.y;
  const int b = bh / NH_, h = bh % NH_, hkv = h / (NH_ / NKV_);
  const int q0 = qt * 16;
  const int nn = lane & 15;
  const int hi = lane >> 4;
  const int mbase = hi * 8;        // rows handled by this lane's accum registers

  __shared__ alignas(16) __bf16 Plds[WPB_ATT][16][32];

  // Q fragments (A-matrix 16x32 layout), K split over 4 chunks of 32.
  F16x16 qf[4];
#pragma unroll
  for (int f = 0; f < 4; ++f) {
    const size_t base = ((size_t)(b * SS + q0 + nn) * NH_ + h) * DD + f * 32 + hi * 8;
    qf[f].q[0] = *(const uint4*)(qb + base);
    qf[f].q[1] = *(const uint4*)(qb + base + 16);
  }

  v8f o[8];
  float mi[8], li[8];
#pragma unroll
  for (int j = 0; j < 8; ++j) o[j] = v8f_zero();
#pragma unroll
  for (int r = 0; r < 8; ++r) { mi[r] = -1e30f; li[r] = 0.f; }

  const float scale = 0.08838834764831845f;   // 1/sqrt(128)

  for (int kv0 = 0; kv0 < q0 + 16; kv0 += 32) {
    // prefetch next key/value block toward the WGP (global_prefetch_b8)
    if (kv0 + 32 < q0 + 16) {
      __builtin_prefetch(kb + ((size_t)(b * SS + kv0 + 32 + nn) * NKV_ + hkv) * DD, 0, 3);
      __builtin_prefetch(vt + ((size_t)((b * NKV_ + hkv) * DD + nn)) * SS + kv0 + 32, 0, 3);
    }
    // ---- scores S = Q @ K^T for 32 keys (two 16-key column tiles)
    v8f sc[2];
    sc[0] = v8f_zero();
    sc[1] = v8f_zero();
#pragma unroll
    for (int f = 0; f < 4; ++f) {
#pragma unroll
      for (int t = 0; t < 2; ++t) {
        F16x16 kf;
        const int j = kv0 + t * 16 + nn;     // key row = B-matrix column
        const size_t base = ((size_t)(b * SS + j) * NKV_ + hkv) * DD + f * 32 + hi * 8;
        kf.q[0] = *(const uint4*)(kb + base);
        kf.q[1] = *(const uint4*)(kb + base + 16);
        sc[t] = __builtin_amdgcn_wmma_f32_16x16x32_bf16(
            false, qf[f].v, false, kf.v, (short)0, sc[t], false, false);
      }
    }
    // ---- online softmax over the 8 rows this lane's registers touch
    float alpha[8];
#pragma unroll
    for (int r = 0; r < 8; ++r) {
      const int row = q0 + mbase + r;
#pragma unroll
      for (int t = 0; t < 2; ++t) {
        const int key = kv0 + t * 16 + nn;
        const float v = sc[t][r] * scale;
        sc[t][r] = (key <= row) ? v : -1e30f;   // causal mask
      }
      float rm = fmaxf(sc[0][r], sc[1][r]);
#pragma unroll
      for (int off = 1; off < 16; off <<= 1) rm = fmaxf(rm, __shfl_xor(rm, off, 32));
      const float mnew = fmaxf(mi[r], rm);
      alpha[r] = __expf(mi[r] - mnew);
      const float p0 = __expf(sc[0][r] - mnew);
      const float p1 = __expf(sc[1][r] - mnew);
      float ps = p0 + p1;
#pragma unroll
      for (int off = 1; off < 16; off <<= 1) ps += __shfl_xor(ps, off, 32);
      li[r] = li[r] * alpha[r] + ps;
      mi[r] = mnew;
      // C-layout (m = mbase+r, n = nn) -> LDS, bf16
      Plds[wave][mbase + r][nn]      = (__bf16)p0;
      Plds[wave][mbase + r][16 + nn] = (__bf16)p1;
    }
    // rescale running output
#pragma unroll
    for (int j = 0; j < 8; ++j)
#pragma unroll
      for (int r = 0; r < 8; ++r) o[j][r] *= alpha[r];

    // reload P in A-matrix 16x32 layout (same-wave LDS bounce, no barrier)
    F16x16 pf;
    pf.q[0] = *(const uint4*)&Plds[wave][nn][hi * 8];
    pf.q[1] = *(const uint4*)&Plds[wave][nn][hi * 8 + 16];

    // ---- O += P @ V  (V columns contiguous thanks to pre-transpose)
#pragma unroll
    for (int j = 0; j < 8; ++j) {
      F16x16 vf;
      const int dcol = j * 16 + nn;
      const size_t base = ((size_t)((b * NKV_ + hkv) * DD + dcol)) * SS + kv0 + hi * 8;
      vf.q[0] = *(const uint4*)(vt + base);
      vf.q[1] = *(const uint4*)(vt + base + 16);
      o[j] = __builtin_amdgcn_wmma_f32_16x16x32_bf16(
          false, pf.v, false, vf.v, (short)0, o[j], false, false);
    }
  }

  // ---- write O / l  as bf16 [b, s, h*D + col]
#pragma unroll
  for (int j = 0; j < 8; ++j)
#pragma unroll
    for (int r = 0; r < 8; ++r) {
      const int row = q0 + mbase + r;
      const int col = h * DD + j * 16 + nn;
      ob[(size_t)(b * SS + row) * (NH_ * DD) + col] = (__bf16)(o[j][r] / li[r]);
    }
}

// ---------------------------------------------------------------------------
// Host orchestration
// ---------------------------------------------------------------------------
extern "C" void kernel_launch(void* const* d_in, const int* in_sizes, int n_in,
                              void* d_out, int out_size, void* d_ws, size_t ws_size,
                              hipStream_t stream) {
  const float* x    = (const float*)d_in[0];
  const int*   pos  = (const int*)  d_in[1];
  const float* wq   = (const float*)d_in[2];
  const float* bq   = (const float*)d_in[3];
  const float* wk   = (const float*)d_in[4];
  const float* bk   = (const float*)d_in[5];
  const float* wv   = (const float*)d_in[6];
  const float* bv   = (const float*)d_in[7];
  const float* wo   = (const float*)d_in[8];
  const float* wg   = (const float*)d_in[9];
  const float* wu   = (const float*)d_in[10];
  const float* wd   = (const float*)d_in[11];
  const float* ln1  = (const float*)d_in[12];
  const float* ln2  = (const float*)d_in[13];
  float* out = (float*)d_out;

  char* cur = (char*)d_ws;
  auto alloc = [&](size_t bytes) {
    void* p = (void*)cur;
    cur += (bytes + 255) & ~(size_t)255;
    return p;
  };

  __bf16* wqkv_t  = (__bf16*)alloc((size_t)NQKV * HH * 2);        // [3072][2048]
  __bf16* wo_t    = (__bf16*)alloc((size_t)HH * HH * 2);          // [2048][2048]
  __bf16* wgate_t = (__bf16*)alloc((size_t)INTER_ * HH * 2);      // [8192][2048]
  __bf16* wup_t   = (__bf16*)alloc((size_t)INTER_ * HH * 2);
  __bf16* wdown_t = (__bf16*)alloc((size_t)HH * INTER_ * 2);      // [2048][8192]
  __bf16* h_b     = (__bf16*)alloc((size_t)MTOT * HH * 2);
  float*  qkv     = (float*) alloc((size_t)MTOT * NQKV * 4);
  __bf16* q_b     = (__bf16*)alloc((size_t)MTOT * NH_ * DD * 2);
  __bf16* k_b     = (__bf16*)alloc((size_t)MTOT * NKV_ * DD * 2);
  __bf16* vt_b    = (__bf16*)alloc((size_t)BB * NKV_ * DD * SS * 2);
  __bf16* attn_b  = (__bf16*)alloc((size_t)MTOT * HH * 2);
  float*  h1      = (float*) alloc((size_t)MTOT * HH * 4);
  __bf16* h2_b    = (__bf16*)alloc((size_t)MTOT * HH * 2);
  __bf16* sg_b    = (__bf16*)alloc((size_t)MTOT * INTER_ * 2);
  __bf16* act_b   = (__bf16*)alloc((size_t)MTOT * INTER_ * 2);

  const dim3 tb(32, 8);
  // QKV weights fused into one [3072][2048] transposed bf16 buffer
  k_transpose_bf16<<<dim3(HH / 32, HH / 32), tb, 0, stream>>>(wq, wqkv_t, HH, HH);
  k_transpose_bf16<<<dim3((NKV_ * DD) / 32, HH / 32), tb, 0, stream>>>(
      wk, wqkv_t + (size_t)(NH_ * DD) * HH, HH, NKV_ * DD);
  k_transpose_bf16<<<dim3((NKV_ * DD) / 32, HH / 32), tb, 0, stream>>>(
      wv, wqkv_t + (size_t)(NH_ * DD + NKV_ * DD) * HH, HH, NKV_ * DD);
  k_transpose_bf16<<<dim3(HH / 32, HH / 32), tb, 0, stream>>>(wo, wo_t, HH, HH);
  k_transpose_bf16<<<dim3(INTER_ / 32, HH / 32), tb, 0, stream>>>(wg, wgate_t, HH, INTER_);
  k_transpose_bf16<<<dim3(INTER_ / 32, HH / 32), tb, 0, stream>>>(wu, wup_t, HH, INTER_);
  k_transpose_bf16<<<dim3(HH / 32, INTER_ / 32), tb, 0, stream>>>(wd, wdown_t, INTER_, HH);

  // 1) RMSNorm -> bf16
  k_rmsnorm<<<MTOT, 256, 0, stream>>>(x, ln1, h_b, HH);
  // 2) fused QKV GEMM (N=3072)
  k_gemm<<<dim3(NQKV / 128, MTOT / 128), 256, 0, stream>>>(
      h_b, wqkv_t, qkv, nullptr, MTOT, NQKV, HH, 0);
  // 3) RoPE(q), RoPE(k), V transpose (+biases)
  k_rope<<<(BB * SS * NH_ * 64 + 255) / 256, 256, 0, stream>>>(
      qkv, bq, pos, q_b, NH_, 0);
  k_rope<<<(BB * SS * NKV_ * 64 + 255) / 256, 256, 0, stream>>>(
      qkv, bk, pos, k_b, NKV_, NH_ * DD);
  k_vtrans<<<(BB * SS * NKV_ * DD + 255) / 256, 256, 0, stream>>>(qkv, bv, vt_b);
  // 4) causal GQA flash attention
  k_attention<<<dim3(SS / 16 / WPB_ATT, BB * NH_), 32 * WPB_ATT, 0, stream>>>(
      q_b, k_b, vt_b, attn_b);
  // 5) O-proj + residual(x) -> h1 (fp32)
  k_gemm<<<dim3(HH / 128, MTOT / 128), 256, 0, stream>>>(
      attn_b, wo_t, h1, x, MTOT, HH, HH, 1);
  // 6) RMSNorm2 -> bf16
  k_rmsnorm<<<MTOT, 256, 0, stream>>>(h1, ln2, h2_b, HH);
  // 7) gate GEMM -> silu -> bf16
  k_gemm<<<dim3(INTER_ / 128, MTOT / 128), 256, 0, stream>>>(
      h2_b, wgate_t, sg_b, nullptr, MTOT, INTER_, HH, 2);
  // 8) up GEMM * silu(gate) -> bf16
  k_gemm<<<dim3(INTER_ / 128, MTOT / 128), 256, 0, stream>>>(
      h2_b, wup_t, act_b, sg_b, MTOT, INTER_, HH, 3);
  // 9) down GEMM + residual(h1) -> d_out (fp32)
  k_gemm<<<dim3(HH / 128, MTOT / 128), 256, 0, stream>>>(
      act_b, wdown_t, out, h1, MTOT, HH, INTER_, 1);
}
